// SelfAttention_17368847745339
// MI455X (gfx1250) — matrix operand order repaired and checked by reading, not compiled, not run on hardware
//
#include <hip/hip_runtime.h>
#include <hip/hip_bf16.h>

typedef __attribute__((ext_vector_type(16))) __bf16 v16bf;
typedef __attribute__((ext_vector_type(8)))  float  v8f;

using bf16 = __hip_bfloat16;

#define HD  1024
#define SEQ 2048
#define NB  4

union FragU { v16bf v; uint4 u[2]; };

// A fragment: 16x32 bf16 (MxK). lane&15 = row. lane<16: K 0..7 & 16..23 ; lane>=16: K 8..15 & 24..31
__device__ __forceinline__ v16bf load_frag_a(const bf16* __restrict__ A, int lda,
                                             int row, int kb, int lane) {
  int r    = row + (lane & 15);
  int koff = (lane >> 4) * 8;
  const bf16* p = A + (size_t)r * lda + kb + koff;
  FragU f;
  f.u[0] = *(const uint4*)(p);
  f.u[1] = *(const uint4*)(p + 16);
  return f.v;
}

// B fragment: 32x16 bf16 (KxN), stored as N rows of K (NT gemm). lane&15 = col. lane<16: K 0..7 & 16..23 ; lane>=16: K 8..15 & 24..31
__device__ __forceinline__ v16bf load_frag_b(const bf16* __restrict__ B, int ldb,
                                             int col, int kb, int lane) {
  int c    = col + (lane & 15);
  int koff = (lane >> 4) * 8;
  const bf16* p = B + (size_t)c * ldb + kb + koff;
  FragU f;
  f.u[0] = *(const uint4*)(p);
  f.u[1] = *(const uint4*)(p + 16);
  return f.v;
}

__device__ __forceinline__ void load_frags(const bf16* __restrict__ A, int lda,
                                           const bf16* __restrict__ B, int ldb,
                                           int row, int col, int kb, int lane,
                                           v16bf a[4], v16bf b[4]) {
#pragma unroll
  for (int i = 0; i < 4; ++i) a[i] = load_frag_a(A, lda, row + 16 * i, kb, lane);
#pragma unroll
  for (int i = 0; i < 4; ++i) b[i] = load_frag_b(B, ldb, col + 16 * i, kb, lane);
}

__device__ __forceinline__ void wmma16(const v16bf a[4], const v16bf b[4], v8f acc[4][4]) {
#pragma unroll
  for (int mi = 0; mi < 4; ++mi)
#pragma unroll
    for (int ni = 0; ni < 4; ++ni)
      acc[mi][ni] = __builtin_amdgcn_wmma_f32_16x16x32_bf16(false, a[mi], false, b[ni],
                                                            (short)0, acc[mi][ni], false, false);
}

// Wave computes a 64(M) x 64(N) tile with double-buffered fragments. K % 64 == 0.
__device__ __forceinline__ void gemm_tile64(const bf16* __restrict__ A, int lda,
                                            const bf16* __restrict__ B, int ldb,
                                            int row, int col, int K, int lane,
                                            v8f acc[4][4]) {
#pragma unroll
  for (int mi = 0; mi < 4; ++mi)
#pragma unroll
    for (int ni = 0; ni < 4; ++ni)
#pragma unroll
      for (int j = 0; j < 8; ++j) acc[mi][ni][j] = 0.0f;

  v16bf a0[4], b0[4], a1[4], b1[4];
  load_frags(A, lda, B, ldb, row, col, 0, lane, a0, b0);
  for (int kb = 0; kb < K; kb += 64) {
    load_frags(A, lda, B, ldb, row, col, kb + 32, lane, a1, b1);
    wmma16(a0, b0, acc);
    if (kb + 64 < K) load_frags(A, lda, B, ldb, row, col, kb + 64, lane, a0, b0);
    wmma16(a1, b1, acc);
  }
}

__global__ void f32_to_bf16_kernel(const float* __restrict__ in, bf16* __restrict__ out, int n) {
  int i = blockIdx.x * blockDim.x + threadIdx.x;
  if (i < n) out[i] = __float2bfloat16(in[i]);
}

// QKV projection: rows = B*S flattened, cols = 3H. Q,K row-major bf16; V stored transposed [b,h,s].
__global__ __launch_bounds__(128) void qkv_kernel(const bf16* __restrict__ X,
                                                  const bf16* __restrict__ W,
                                                  const float* __restrict__ bias,
                                                  bf16* __restrict__ Qb,
                                                  bf16* __restrict__ Kb,
                                                  bf16* __restrict__ Vt) {
  int lane = threadIdx.x & 31, wave = threadIdx.x >> 5;
  int row = blockIdx.x * 128 + (wave >> 1) * 64;
  int col = blockIdx.y * 128 + (wave & 1) * 64;
  v8f acc[4][4];
  gemm_tile64(X, HD, W, HD, row, col, HD, lane, acc);

  int m0 = (lane >> 4) * 8;
#pragma unroll
  for (int ni = 0; ni < 4; ++ni) {
    int nc  = col + 16 * ni + (lane & 15);
    float bv = bias[nc];
    int seg = nc >> 10, nl = nc & (HD - 1);
#pragma unroll
    for (int mi = 0; mi < 4; ++mi) {
#pragma unroll
      for (int j = 0; j < 8; ++j) {
        int m = row + 16 * mi + m0 + j;       // global row in [0, B*S)
        bf16 h = __float2bfloat16(acc[mi][ni][j] + bv);
        if (seg == 0)      Qb[(size_t)m * HD + nl] = h;
        else if (seg == 1) Kb[(size_t)m * HD + nl] = h;
        else {
          int bidx = m >> 11, s = m & (SEQ - 1);
          Vt[((size_t)bidx * HD + nl) * SEQ + s] = h;
        }
      }
    }
  }
}

// scores[b] = Q[b] @ K[b]^T * 1/sqrt(H), bf16 out
__global__ __launch_bounds__(128) void score_kernel(const bf16* __restrict__ Qb,
                                                    const bf16* __restrict__ Kb,
                                                    bf16* __restrict__ Sc) {
  int lane = threadIdx.x & 31, wave = threadIdx.x >> 5;
  int b = blockIdx.z;
  const bf16* A = Qb + (size_t)b * SEQ * HD;
  const bf16* B = Kb + (size_t)b * SEQ * HD;
  bf16* C = Sc + (size_t)b * SEQ * SEQ;
  int row = blockIdx.x * 128 + (wave >> 1) * 64;
  int col = blockIdx.y * 128 + (wave & 1) * 64;
  v8f acc[4][4];
  gemm_tile64(A, HD, B, HD, row, col, HD, lane, acc);

  const float scale = 0.03125f;  // 1/sqrt(1024)
  int m0 = (lane >> 4) * 8;
#pragma unroll
  for (int ni = 0; ni < 4; ++ni) {
    int nc = col + 16 * ni + (lane & 15);
#pragma unroll
    for (int mi = 0; mi < 4; ++mi)
#pragma unroll
      for (int j = 0; j < 8; ++j)
        C[(size_t)(row + 16 * mi + m0 + j) * SEQ + nc] =
            __float2bfloat16(acc[mi][ni][j] * scale);
  }
}

// in-place row softmax over bf16 scores, one 256-thread block per row of length 2048
__global__ __launch_bounds__(256) void softmax_kernel(bf16* __restrict__ Sc) {
  bf16* rowp = Sc + (size_t)blockIdx.x * SEQ;
  int tid = threadIdx.x;
  float v[8];
  float mx = -INFINITY;
#pragma unroll
  for (int j = 0; j < 8; ++j) {
    v[j] = __bfloat162float(rowp[tid * 8 + j]);
    mx = fmaxf(mx, v[j]);
  }
  __shared__ float red[256];
  red[tid] = mx;
  __syncthreads();
  for (int s = 128; s > 0; s >>= 1) {
    if (tid < s) red[tid] = fmaxf(red[tid], red[tid + s]);
    __syncthreads();
  }
  mx = red[0];
  __syncthreads();
  float sum = 0.0f;
#pragma unroll
  for (int j = 0; j < 8; ++j) {
    v[j] = __expf(v[j] - mx);
    sum += v[j];
  }
  red[tid] = sum;
  __syncthreads();
  for (int s = 128; s > 0; s >>= 1) {
    if (tid < s) red[tid] += red[tid + s];
    __syncthreads();
  }
  float inv = 1.0f / red[0];
#pragma unroll
  for (int j = 0; j < 8; ++j)
    rowp[tid * 8 + j] = __float2bfloat16(v[j] * inv);
}

// out[b] = P[b] @ V[b]  (P bf16 row-major SxS, V pre-transposed Vt[b,h,s] so NT loader works), f32 out
__global__ __launch_bounds__(128) void av_kernel(const bf16* __restrict__ P,
                                                 const bf16* __restrict__ Vt,
                                                 float* __restrict__ Out) {
  int lane = threadIdx.x & 31, wave = threadIdx.x >> 5;
  int b = blockIdx.z;
  const bf16* A = P  + (size_t)b * SEQ * SEQ;  // lda = SEQ
  const bf16* B = Vt + (size_t)b * HD * SEQ;   // ldb = SEQ (row h holds V[:,h] over s)
  float* C = Out + (size_t)b * SEQ * HD;
  int row = blockIdx.x * 128 + (wave >> 1) * 64;
  int col = blockIdx.y * 128 + (wave & 1) * 64;
  v8f acc[4][4];
  gemm_tile64(A, SEQ, B, SEQ, row, col, SEQ, lane, acc);

  int m0 = (lane >> 4) * 8;
#pragma unroll
  for (int ni = 0; ni < 4; ++ni) {
    int nc = col + 16 * ni + (lane & 15);
#pragma unroll
    for (int mi = 0; mi < 4; ++mi)
#pragma unroll
      for (int j = 0; j < 8; ++j)
        C[(size_t)(row + 16 * mi + m0 + j) * HD + nc] = acc[mi][ni][j];
  }
}

extern "C" void kernel_launch(void* const* d_in, const int* in_sizes, int n_in,
                              void* d_out, int out_size, void* d_ws, size_t ws_size,
                              hipStream_t stream) {
  (void)in_sizes; (void)n_in; (void)out_size; (void)ws_size;
  const float* x    = (const float*)d_in[0];
  const float* W    = (const float*)d_in[1];
  const float* bias = (const float*)d_in[2];
  float* out = (float*)d_out;

  char* ws = (char*)d_ws;
  // workspace layout (bytes), all 256-aligned:
  bf16* xb = (bf16*)(ws + 0);          // 16 MB : x as bf16
  bf16* wb = (bf16*)(ws + 16777216);   //  6 MB : W as bf16
  bf16* qb = (bf16*)(ws + 23068672);   // 16 MB : Q bf16 [B*S, H]
  bf16* kb = (bf16*)(ws + 39845888);   // 16 MB : K bf16 [B*S, H]
  bf16* vt = (bf16*)(ws + 56623104);   // 16 MB : V^T bf16 [B, H, S]
  bf16* sc = (bf16*)(ws + 73400320);   // 32 MB : scores / probs bf16 [B, S, S]

  const int nx = NB * SEQ * HD;   // 8388608
  const int nw = 3 * HD * HD;     // 3145728

  f32_to_bf16_kernel<<<nx / 256, 256, 0, stream>>>(x, xb, nx);
  f32_to_bf16_kernel<<<nw / 256, 256, 0, stream>>>(W, wb, nw);

  qkv_kernel  <<<dim3(NB * SEQ / 128, (3 * HD) / 128, 1), 128, 0, stream>>>(xb, wb, bias, qb, kb, vt);
  score_kernel<<<dim3(SEQ / 128, SEQ / 128, NB),          128, 0, stream>>>(qb, kb, sc);
  softmax_kernel<<<NB * SEQ, 256, 0, stream>>>(sc);
  av_kernel   <<<dim3(SEQ / 128, HD / 128, NB),           128, 0, stream>>>(sc, vt, out);
}
